// VRWKV_SpatialMix_V6_37297495998786
// MI455X (gfx1250) — matrix operand order repaired and checked by reading, not compiled, not run on hardware
//
#include <hip/hip_runtime.h>
#include <hip/hip_bf16.h>
#include <hip/hip_fp16.h>

typedef __attribute__((ext_vector_type(16))) _Float16 v16h;
typedef __attribute__((ext_vector_type(8)))  _Float16 v8h;
typedef __attribute__((ext_vector_type(2)))  _Float16 v2h;
typedef __attribute__((ext_vector_type(8)))  float    v8f;
typedef __attribute__((ext_vector_type(8)))  float    v8fv;

#define CBATCH 8
#define CT     4096
#define CCH    192
#define CNH    24
#define CHS    8
#define CMIX   32
#define CDEC   64
#define CM     (CBATCH*CT)   // 32768 tokens

// ---------------------------------------------------------------------------
// WMMA 16x16x32 f16 fragment conventions used in this file (contiguous k per lane half, not the ISA table order):
//  A: lane lo=lane&15 -> M=lo; halves = A[row, k0 + hi*16 .. +15]
//     -> two contiguous 32B f32 runs of one row, converted to f16.
//  B: lane -> N = lane&15, K = (lane>>4)*16 + h. We PRE-SWIZZLE weights in global
//     memory to Bs[kb][nt][lane][16h] so a fragment is ONE contiguous 32B/lane load.
//  C/D: VGPR r -> M = (lane>>4)*8 + r, N = lane&15.
// ---------------------------------------------------------------------------

// ====================== weight-prep kernels ======================

// Fold the 4 diff-convs into one 3x3 kernel, written directly in swizzled
// fragment layout: Bconv[tap][kb(6)][nt(12)][lane(32)][h(16)], k=cin, n=cout.
__global__ __launch_bounds__(256)
void prep_conv_w(const float* __restrict__ w_lp, const float* __restrict__ w_hd,
                 const float* __restrict__ w_vd, const float* __restrict__ w_cd,
                 const float* __restrict__ theta, _Float16* __restrict__ Bconv)
{
    int idx = blockIdx.x * blockDim.x + threadIdx.x;
    if (idx >= CCH * CCH) return;
    int co = idx / CCH, ci = idx % CCH;
    float lp = w_lp[co * CCH + ci];
    float hd[3], vd[3], cd[9], k[9];
    float cdsum = 0.f;
    #pragma unroll
    for (int q = 0; q < 3; ++q) {
        hd[q] = w_hd[(co * CCH + ci) * 3 + q];
        vd[q] = w_vd[(co * CCH + ci) * 3 + q];
    }
    #pragma unroll
    for (int p = 0; p < 9; ++p) { cd[p] = w_cd[(co * CCH + ci) * 9 + p]; cdsum += cd[p]; }
    #pragma unroll
    for (int p = 0; p < 9; ++p) k[p] = cd[p];
    k[1] += lp; k[3] += lp; k[5] += lp; k[7] += lp; k[4] += -4.f * lp;   // laplacian
    k[0] += hd[0]; k[3] += hd[1]; k[6] += hd[2];                        // horiz diff
    k[2] -= hd[0]; k[5] -= hd[1]; k[8] -= hd[2];
    k[0] += vd[0]; k[1] += vd[1]; k[2] += vd[2];                        // vert diff
    k[6] -= vd[0]; k[7] -= vd[1]; k[8] -= vd[2];
    k[4] += -theta[0] * cdsum;                                          // cd 1x1
    int kb = ci >> 5, hi = (ci >> 4) & 1, h = ci & 15;
    int nt = co >> 4, lo = co & 15;
    int lane = hi * 16 + lo;
    #pragma unroll
    for (int p = 0; p < 9; ++p)
        Bconv[(((((size_t)p * 6 + kb) * 12 + nt) * 32) + lane) * 16 + h] = (_Float16)k[p];
}

// Generic weight swizzle: W[K][N] f32 -> fragment-native f16.
__global__ __launch_bounds__(256)
void swizzle_w_kernel(const float* __restrict__ W, _Float16* __restrict__ out,
                      int K, int N)
{
    int idx = blockIdx.x * blockDim.x + threadIdx.x;
    if (idx >= K * N) return;
    int k = idx / N, n = idx % N;
    int kb = k >> 5, hi = (k >> 4) & 1, h = k & 15;
    int nt = n >> 4, lo = n & 15;
    int lane = hi * 16 + lo;
    int NT = N >> 4;
    out[((((size_t)kb * NT + nt) * 32) + lane) * 16 + h] = (_Float16)W[idx];
}

// ====================== implicit-GEMM 3x3 conv (WMMA) ======================
// Block: one image row (64 pixels) x 64 cout. LDS only for the padded halo.
__global__ __launch_bounds__(256)
void conv_mix_kernel(const float* __restrict__ x, const _Float16* __restrict__ Bconv,
                     const float* __restrict__ maa_x,
                     float* __restrict__ xxo, float* __restrict__ xxxo)
{
    __shared__ __align__(16) _Float16 sA[3 * 66 * 32]; // 3 rows x 66 pix x 32 ch
    const int ntb = blockIdx.x * 4;          // n0 = ntb*16
    const int row = blockIdx.y;              // 0..B*64-1
    const int b = row >> 6, yy = row & 63;
    const int tid = threadIdx.x;
    const int wave = tid >> 5, lane = tid & 31;
    const int lo = lane & 15, hi = lane >> 4;
    const int mi = wave & 3;                 // pixel sub-tile
    const int jb = (wave >> 2) * 2;          // col sub-tiles jb, jb+1
    v8f acc[2] = {};
    const size_t xbase = (size_t)b * CT * CCH;
    const v16h* Bf = (const v16h*)Bconv;

    for (int k0 = 0, kb = 0; k0 < CCH; k0 += 32, ++kb) {
        __syncthreads();
        for (int idx = tid; idx < 3 * 66 * 16; idx += 256) {
            int cp = idx & 15, p = (idx >> 4) % 66, rr = idx / (66 * 16);
            int gy = yy + rr - 1, gx = p - 1;
            v2h hv = {(_Float16)0.f, (_Float16)0.f};
            if (gy >= 0 && gy < 64 && gx >= 0 && gx < 64) {
                const float2 f = *(const float2*)&x[xbase + (size_t)(gy * 64 + gx) * CCH
                                                   + k0 + 2 * cp];
                hv[0] = (_Float16)f.x; hv[1] = (_Float16)f.y;
            }
            *(v2h*)(sA + rr * (66 * 32) + p * 32 + 2 * cp) = hv;
        }
        __syncthreads();
        #pragma unroll
        for (int tap = 0; tap < 9; ++tap) {
            const int dy = tap / 3, dx = tap % 3;
            const _Float16* ap = sA + dy * (66 * 32) + (mi * 16 + lo + dx) * 32;
            v8h a0 = *(const v8h*)(ap + hi * 16);
            v8h a1 = *(const v8h*)(ap + hi * 16 + 8);
            v16h af = __builtin_shufflevector(a0, a1, 0, 1, 2, 3, 4, 5, 6, 7,
                                              8, 9, 10, 11, 12, 13, 14, 15);
            #pragma unroll
            for (int q = 0; q < 2; ++q) {
                v16h bf = Bf[((((size_t)tap * 6 + kb) * 12) + ntb + jb + q) * 32 + lane];
                acc[q] = __builtin_amdgcn_wmma_f32_16x16x32_f16(
                    false, af, false, bf, (short)0, acc[q], false, false);
            }
        }
    }
    // epilogue: xx = conv - x ; xxx = x + xx * maa_x
    #pragma unroll
    for (int q = 0; q < 2; ++q) {
        int gn = (ntb + jb + q) * 16 + lo;
        #pragma unroll
        for (int r = 0; r < 8; ++r) {
            int pix = mi * 16 + hi * 8 + r;
            size_t off = xbase + (size_t)(yy * 64 + pix) * CCH + gn;
            float xv = x[off];
            float d = acc[q][r] - xv;
            xxo[off]  = d;
            xxxo[off] = xv + d * maa_x[gn];
        }
    }
}

// ====================== LDS-free WMMA GEMM with fused epilogues ======================
// Out[M,N] = epilogue(A[M,K(+acol)] @ Bs);  block tile 128x64, 8 waves of 16x64.
// A fragments: direct global 32B vector loads of one row per lane, cvt to f16.
// B fragments: one contiguous 32B/lane load from pre-swizzled weights (L2 resident).
// EP: 0=none 1=tanh 2=silu 3=mix(x+xx*(bias+acc)) 4=decay(exp(-exp(bias+acc)))
template <int EP>
__global__ __launch_bounds__(256)
void gemm_ep_kernel(const float* __restrict__ A, int lda, int acol,
                    const _Float16* __restrict__ Bs,
                    float* __restrict__ Out, int ldo,
                    int M, int N, int K,
                    const float* __restrict__ xg, const float* __restrict__ xxg,
                    const float* __restrict__ bias)
{
    const int tid = threadIdx.x;
    const int wave = tid >> 5, lane = tid & 31;
    const int lo = lane & 15, hi = lane >> 4;
    const int NT = N >> 4;
    const int ntb = blockIdx.x * 4;
    const int m0 = blockIdx.y * 128;
    int gma = m0 + wave * 16 + lo;           // row this lane feeds into A frags
    if (gma >= M) gma = M - 1;
    const float* Ar = A + (size_t)gma * lda + acol;
    const v16h* Bf = (const v16h*)Bs;
    v8f acc[4] = {};
    for (int k0 = 0, kb = 0; k0 < K; k0 += 32, ++kb) {
        v8fv a0 = *(const v8fv*)(Ar + k0 + hi * 16);
        v8fv a1 = *(const v8fv*)(Ar + k0 + hi * 16 + 8);
        v8h h0 = __builtin_convertvector(a0, v8h);
        v8h h1 = __builtin_convertvector(a1, v8h);
        v16h af = __builtin_shufflevector(h0, h1, 0, 1, 2, 3, 4, 5, 6, 7,
                                          8, 9, 10, 11, 12, 13, 14, 15);
        #pragma unroll
        for (int jn = 0; jn < 4; ++jn) {
            v16h bf = Bf[((size_t)kb * NT + ntb + jn) * 32 + lane];
            acc[jn] = __builtin_amdgcn_wmma_f32_16x16x32_f16(
                false, af, false, bf, (short)0, acc[jn], false, false);
        }
    }
    #pragma unroll
    for (int jn = 0; jn < 4; ++jn) {
        int gn = (ntb + jn) * 16 + lo;
        if (gn >= N) continue;
        #pragma unroll
        for (int r = 0; r < 8; ++r) {
            int gm = m0 + wave * 16 + hi * 8 + r;
            if (gm >= M) continue;
            float a = acc[jn][r];
            float o;
            if (EP == 0)      o = a;
            else if (EP == 1) o = tanhf(a);
            else if (EP == 2) o = a / (1.f + expf(-a));                        // silu
            else if (EP == 3) o = xg[(size_t)gm * CCH + gn]
                                + xxg[(size_t)gm * CCH + gn] * (bias[gn] + a); // mix
            else              o = expf(-expf(bias[gn] + a));                   // decay
            Out[(size_t)gm * ldo + gn] = o;
        }
    }
}

// ====================== WKV6 serial scan: one wave per (b, head) ======================
__global__ __launch_bounds__(32)
void wkv_scan_kernel(const float* __restrict__ r, const float* __restrict__ k,
                     const float* __restrict__ v, const float* __restrict__ dcy,
                     const float* __restrict__ u, float* __restrict__ y)
{
    const int bh = blockIdx.x;            // 0..B*NH-1
    const int b = bh / CNH, h = bh % CNH;
    const int lane = threadIdx.x;
    const int i  = lane & 7;
    const int j0 = lane >> 3;             // 0..3
    const int grp = lane >> 3, li = lane & 7;
    const float u0 = u[h * CHS + j0];
    const float u1 = u[h * CHS + j0 + 4];
    const size_t base = (size_t)b * CT * CCH + h * CHS;
    const float* src = (grp == 0) ? r : (grp == 1) ? k : (grp == 2) ? v : dcy;
    float S0 = 0.f, S1 = 0.f;
    float cur = src[base + li];
    for (int t = 0; t < CT; ++t) {
        float nxt = 0.f;
        if (t + 1 < CT) nxt = src[base + (size_t)(t + 1) * CCH + li];
        if (t + 8 < CT) __builtin_prefetch(src + base + (size_t)(t + 8) * CCH + li, 0, 0);
        float rv0 = __shfl(cur, j0, 32);
        float rv1 = __shfl(cur, j0 + 4, 32);
        float kv0 = __shfl(cur, 8 + j0, 32);
        float kv1 = __shfl(cur, 8 + j0 + 4, 32);
        float vv  = __shfl(cur, 16 + i, 32);
        float dv0 = __shfl(cur, 24 + j0, 32);
        float dv1 = __shfl(cur, 24 + j0 + 4, 32);
        float kvi0 = kv0 * vv, kvi1 = kv1 * vv;
        float c = rv0 * (S0 + u0 * kvi0) + rv1 * (S1 + u1 * kvi1);
        c += __shfl_xor(c, 8, 32);
        c += __shfl_xor(c, 16, 32);
        if (lane < 8) y[base + (size_t)t * CCH + lane] = c;
        S0 = dv0 * S0 + kvi0;
        S1 = dv1 * S1 + kvi1;
        cur = nxt;
    }
}

// ====================== LayerNorm + gate ======================
__global__ __launch_bounds__(256)
void ln_gate_kernel(const float* __restrict__ y, const float* __restrict__ g,
                    const float* __restrict__ lng, const float* __restrict__ lnb,
                    float* __restrict__ z)
{
    int tok = blockIdx.x * 8 + (threadIdx.x >> 5);
    int lane = threadIdx.x & 31;
    if (tok >= CM) return;
    const float* yr = y + (size_t)tok * CCH;
    float vals[6], s = 0.f;
    #pragma unroll
    for (int q = 0; q < 6; ++q) { vals[q] = yr[lane + 32 * q]; s += vals[q]; }
    #pragma unroll
    for (int o = 16; o > 0; o >>= 1) s += __shfl_xor(s, o, 32);
    float mu = s * (1.f / CCH);
    float v2 = 0.f;
    #pragma unroll
    for (int q = 0; q < 6; ++q) { float d = vals[q] - mu; v2 += d * d; }
    #pragma unroll
    for (int o = 16; o > 0; o >>= 1) v2 += __shfl_xor(v2, o, 32);
    float rstd = rsqrtf(v2 * (1.f / CCH) + 1e-5f);
    const float* gr = g + (size_t)tok * CCH;
    #pragma unroll
    for (int q = 0; q < 6; ++q) {
        int c = lane + 32 * q;
        z[(size_t)tok * CCH + c] = ((vals[q] - mu) * rstd * lng[c] + lnb[c]) * gr[c];
    }
}

// ====================== host side ======================

static inline void* ws_take(char*& cur, size_t bytes)
{
    void* p = (void*)cur;
    cur += (bytes + 255) & ~(size_t)255;
    return p;
}

static void launch_gemm(int ep, const float* A, int lda, int acol, const _Float16* Bs,
                        float* Out, int ldo, int M, int N, int K,
                        const float* xg, const float* xxg, const float* bias,
                        hipStream_t s)
{
    dim3 g((N + 63) / 64, (M + 127) / 128), b(256);
    switch (ep) {
    case 0: gemm_ep_kernel<0><<<g, b, 0, s>>>(A, lda, acol, Bs, Out, ldo, M, N, K, xg, xxg, bias); break;
    case 1: gemm_ep_kernel<1><<<g, b, 0, s>>>(A, lda, acol, Bs, Out, ldo, M, N, K, xg, xxg, bias); break;
    case 2: gemm_ep_kernel<2><<<g, b, 0, s>>>(A, lda, acol, Bs, Out, ldo, M, N, K, xg, xxg, bias); break;
    case 3: gemm_ep_kernel<3><<<g, b, 0, s>>>(A, lda, acol, Bs, Out, ldo, M, N, K, xg, xxg, bias); break;
    default: gemm_ep_kernel<4><<<g, b, 0, s>>>(A, lda, acol, Bs, Out, ldo, M, N, K, xg, xxg, bias); break;
    }
}

extern "C" void kernel_launch(void* const* d_in, const int* in_sizes, int n_in,
                              void* d_out, int out_size, void* d_ws, size_t ws_size,
                              hipStream_t stream)
{
    (void)in_sizes; (void)n_in; (void)out_size; (void)ws_size;
    const float* x        = (const float*)d_in[0];
    const float* maa_x    = (const float*)d_in[1];
    const float* maa_bias[5] = { (const float*)d_in[2], (const float*)d_in[3],
                                 (const float*)d_in[4], (const float*)d_in[5],
                                 (const float*)d_in[6] };          // w,k,v,r,g
    const float* maa_w1   = (const float*)d_in[7];
    const float* maa_w2   = (const float*)d_in[8];
    const float* decay_w1 = (const float*)d_in[9];
    const float* decay_w2 = (const float*)d_in[10];
    const float* sp_decay = (const float*)d_in[11];
    const float* u        = (const float*)d_in[12];
    const float* Wmat[5]  = { (const float*)d_in[13], (const float*)d_in[14],
                              (const float*)d_in[15], (const float*)d_in[16],
                              (const float*)d_in[17] };            // Wr,Wk,Wv,Wg,Wo
    const float* ln_g     = (const float*)d_in[18];
    const float* ln_b     = (const float*)d_in[19];
    const float* w_lp     = (const float*)d_in[20];
    const float* w_hd     = (const float*)d_in[21];
    const float* w_vd     = (const float*)d_in[22];
    const float* w_cd     = (const float*)d_in[23];
    const float* theta    = (const float*)d_in[24];
    float* out = (float*)d_out;

    char* cur = (char*)d_ws;
    // swizzled f16 weights
    _Float16* Bconv = (_Float16*)ws_take(cur, (size_t)9 * CCH * CCH * 2);
    _Float16* w1s   = (_Float16*)ws_take(cur, (size_t)CCH * 160 * 2);
    _Float16* w2s   = (_Float16*)ws_take(cur, (size_t)5 * CMIX * CCH * 2);
    _Float16* dw1s  = (_Float16*)ws_take(cur, (size_t)CCH * CDEC * 2);
    _Float16* dw2s  = (_Float16*)ws_take(cur, (size_t)CDEC * CCH * 2);
    _Float16* Wsf[5];
    for (int q = 0; q < 5; ++q) Wsf[q] = (_Float16*)ws_take(cur, (size_t)CCH * CCH * 2);
    // f32 activations
    const size_t MC = (size_t)CM * CCH;
    float* xxb  = (float*)ws_take(cur, MC * 4);
    float* xxxb = (float*)ws_take(cur, MC * 4);   // reused as z after H is consumed
    float* Hb   = (float*)ws_take(cur, (size_t)CM * 160 * 4);
    float* xmix[5];                                // xw,xk,xv,xr,xg
    for (int q = 0; q < 5; ++q) xmix[q] = (float*)ws_take(cur, MC * 4);
    float* rb   = (float*)ws_take(cur, MC * 4);
    float* kb   = (float*)ws_take(cur, MC * 4);
    float* vb   = (float*)ws_take(cur, MC * 4);
    float* gb   = (float*)ws_take(cur, MC * 4);
    float* Hd   = (float*)ws_take(cur, (size_t)CM * CDEC * 4);
    float* dcy  = (float*)ws_take(cur, MC * 4);
    float* yb   = (float*)ws_take(cur, MC * 4);
    float* zb   = xxxb;                            // alias: xxx dead after GEMM1

    // 1) fold diff-convs into one swizzled 3x3 f16 kernel
    prep_conv_w<<<(CCH * CCH + 255) / 256, 256, 0, stream>>>(w_lp, w_hd, w_vd, w_cd,
                                                             theta, Bconv);
    // 2) swizzle GEMM weights to fragment-native f16
    auto swz = [&](const float* s, _Float16* d, int K, int N) {
        swizzle_w_kernel<<<(K * N + 255) / 256, 256, 0, stream>>>(s, d, K, N);
    };
    swz(maa_w1, w1s, CCH, 160);
    for (int f = 0; f < 5; ++f)
        swz(maa_w2 + (size_t)f * CMIX * CCH, w2s + (size_t)f * CMIX * CCH, CMIX, CCH);
    swz(decay_w1, dw1s, CCH, CDEC);
    swz(decay_w2, dw2s, CDEC, CCH);
    for (int q = 0; q < 5; ++q) swz(Wmat[q], Wsf[q], CCH, CCH);

    // 3) implicit-GEMM 3x3 conv -> xx and xxx (fused)
    conv_mix_kernel<<<dim3(3, CBATCH * 64), 256, 0, stream>>>(x, Bconv, maa_x, xxb, xxxb);

    // 4) H = tanh(xxx @ maa_w1)  [M,160]
    launch_gemm(1, xxxb, CCH, 0, w1s, Hb, 160, CM, 160, CCH, nullptr, nullptr, nullptr, stream);

    // 5) xw..xg = x + xx*(maa_f + H_f @ maa_w2[f])   (5 fused GEMMs, K=32)
    for (int f = 0; f < 5; ++f)
        launch_gemm(3, Hb, 160, f * CMIX, w2s + (size_t)f * CMIX * CCH,
                    xmix[f], CCH, CM, CCH, CMIX, x, xxb, maa_bias[f], stream);

    // 6) projections: r=xr@Wr, k=xk@Wk, v=xv@Wv, g=silu(xg@Wg)
    launch_gemm(0, xmix[3], CCH, 0, Wsf[0], rb, CCH, CM, CCH, CCH, nullptr, nullptr, nullptr, stream);
    launch_gemm(0, xmix[1], CCH, 0, Wsf[1], kb, CCH, CM, CCH, CCH, nullptr, nullptr, nullptr, stream);
    launch_gemm(0, xmix[2], CCH, 0, Wsf[2], vb, CCH, CM, CCH, CCH, nullptr, nullptr, nullptr, stream);
    launch_gemm(2, xmix[4], CCH, 0, Wsf[3], gb, CCH, CM, CCH, CCH, nullptr, nullptr, nullptr, stream);

    // 7) decay: Hd = tanh(xw @ decay_w1); dcy = exp(-exp(sd + Hd @ decay_w2))
    launch_gemm(1, xmix[0], CCH, 0, dw1s, Hd, CDEC, CM, CDEC, CCH, nullptr, nullptr, nullptr, stream);
    launch_gemm(4, Hd, CDEC, 0, dw2s, dcy, CCH, CM, CCH, CDEC, nullptr, nullptr, sp_decay, stream);

    // 8) serial WKV6 scan: one wave32 per (b, head)
    wkv_scan_kernel<<<CBATCH * CNH, 32, 0, stream>>>(rb, kb, vb, dcy, u, yb);

    // 9) LayerNorm + gate
    ln_gate_kernel<<<(CM + 7) / 8, 256, 0, stream>>>(yb, gb, ln_g, ln_b, zb);

    // 10) out = z @ Wo
    launch_gemm(0, zb, CCH, 0, Wsf[4], out, CCH, CM, CCH, CCH, nullptr, nullptr, nullptr, stream);
}
